// GraphClassifier_65506841199134
// MI455X (gfx1250) — matrix operand order repaired and checked by reading, not compiled, not run on hardware
//
#include <hip/hip_runtime.h>

#define NN 50000
#define NE 800000
#define DIM 128
#define NG 64
#define NC 10

typedef __attribute__((ext_vector_type(16))) __bf16 v16bf;
typedef __attribute__((ext_vector_type(8)))  float  v8f;
typedef __attribute__((ext_vector_type(4)))  unsigned int u32x4;

union Frag { v16bf bf; u32x4 q[2]; };
union Acc  { v8f v; float f[8]; };

__device__ __forceinline__ unsigned short f2bf(float x) {
  unsigned int u = __builtin_bit_cast(unsigned int, x);
  u += 0x7FFFu + ((u >> 16) & 1u);            // round-to-nearest-even
  return (unsigned short)(u >> 16);
}

// C[M x 128] = op(A[M x 128]) @ W[128 x 128] + bias ; op = relu if relu_in
__global__ __launch_bounds__(256) void gemm128(const float* __restrict__ A,
                                               const float* __restrict__ W,
                                               const float* __restrict__ bias,
                                               float* __restrict__ C,
                                               int M, int relu_in) {
  __shared__ unsigned short sW[DIM * DIM];    // transposed: sW[n*128 + k], bf16 bits
  __shared__ unsigned short sA[64 * DIM];     // 64-row A tile, bf16 bits

  const int tid  = threadIdx.x;
  const int lane = tid & 31;
  const int wave = tid >> 5;
  const int row0blk = blockIdx.x * 64;

  // stage weights transposed (global reads coalesced)
  for (int idx = tid; idx < DIM * DIM; idx += 256) {
    int k = idx >> 7, n = idx & 127;
    sW[n * DIM + k] = f2bf(W[idx]);
  }
  // stage A tile (clamp OOB rows; garbage rows never stored)
  for (int idx = tid; idx < 64 * DIM; idx += 256) {
    int r = idx >> 7, k = idx & 127;
    int row = row0blk + r; if (row >= M) row = M - 1;
    float v = A[(size_t)row * DIM + k];
    if (relu_in) v = fmaxf(v, 0.0f);
    sA[idx] = f2bf(v);
  }
  __syncthreads();

  const int rowgrp  = wave >> 1;              // 0..3 -> 16-row slice
  const int colbase = (wave & 1) * 64;        // 0 or 64
  const int m     = lane & 15;
  const int khalf = lane >> 4;

  Acc acc[4];
  #pragma unroll
  for (int ct = 0; ct < 4; ++ct)
    #pragma unroll
    for (int j = 0; j < 8; ++j) acc[ct].f[j] = 0.0f;

  #pragma unroll
  for (int k0 = 0; k0 < DIM; k0 += 32) {
    // A fragment: lanes 0-15 carry K {0..7,16..23}, lanes 16-31 K {8..15,24..31}
    Frag a;
    const unsigned short* ap = &sA[(rowgrp * 16 + m) * DIM + k0 + khalf * 8];
    a.q[0] = *(const u32x4*)ap;               // K = k0+kh*8 .. +7   -> VGPR0..3
    a.q[1] = *(const u32x4*)(ap + 16);        // K = k0+16+kh*8 ..+7 -> VGPR4..7
    #pragma unroll
    for (int ct = 0; ct < 4; ++ct) {
      // B fragment: col = lane%16; lanes 0-15 K {0..7,16..23}, lanes 16-31 K {8..15,24..31}
      const int n = colbase + ct * 16 + m;
      Frag b;
      const unsigned short* bp = &sW[n * DIM + k0 + khalf * 8];
      b.q[0] = *(const u32x4*)bp;
      b.q[1] = *(const u32x4*)(bp + 16);
      acc[ct].v = __builtin_amdgcn_wmma_f32_16x16x32_bf16(
          false, a.bf, false, b.bf, (short)0, acc[ct].v, false, false);
    }
  }

  // C/D layout: VGPR r -> M = r + 8*(lane>=16), N = lane%16
  const int rowbase = row0blk + rowgrp * 16 + khalf * 8;
  #pragma unroll
  for (int ct = 0; ct < 4; ++ct) {
    const int col = colbase + ct * 16 + m;
    const float bv = bias[col];
    #pragma unroll
    for (int r = 0; r < 8; ++r) {
      const int row = rowbase + r;
      if (row < M) C[(size_t)row * DIM + col] = acc[ct].f[r] + bv;
    }
  }
}

// agg[dst] += h[src] per edge; one wave per edge, lane handles 4 feats
__global__ void edge_scatter(const float* __restrict__ h,
                             const int* __restrict__ ei,
                             float* __restrict__ agg, int n_edges) {
  long long gid = (long long)blockIdx.x * blockDim.x + threadIdx.x;
  int edge = (int)(gid >> 5);
  int lane = (int)(gid & 31);
  if (edge >= n_edges) return;
  int s = ei[edge];
  int d = ei[n_edges + edge];
  const float4 v = *(const float4*)&h[(size_t)s * DIM + lane * 4];
  float* base = &agg[(size_t)d * DIM + lane * 4];
  atomicAdd(base + 0, v.x);
  atomicAdd(base + 1, v.y);
  atomicAdd(base + 2, v.z);
  atomicAdd(base + 3, v.w);
}

__global__ void zero_f32(float* p, int n) {
  int i = blockIdx.x * blockDim.x + threadIdx.x;
  if (i < n) p[i] = 0.0f;
}

// per-feature sum / sum-of-squares (blockDim = 128, thread = feature, coalesced)
__global__ void bn_stats(const float* __restrict__ x, float* sums, float* sumsq, int n_rows) {
  int f = threadIdx.x;
  float s = 0.0f, ss = 0.0f;
  for (int r = blockIdx.x; r < n_rows; r += gridDim.x) {
    float v = x[(size_t)r * DIM + f];
    s += v; ss += v * v;
  }
  atomicAdd(&sums[f], s);
  atomicAdd(&sumsq[f], ss);
}

__global__ void bn_apply_relu(float* __restrict__ x, const float* __restrict__ sums,
                              const float* __restrict__ sumsq, int n_rows) {
  long long i = (long long)blockIdx.x * blockDim.x + threadIdx.x;
  if (i >= (long long)n_rows * DIM) return;
  int f = (int)(i & (DIM - 1));
  const float inv_n = 1.0f / (float)n_rows;
  float mu  = sums[f] * inv_n;
  float var = sumsq[f] * inv_n - mu * mu;
  float sc  = rsqrtf(var + 1e-5f);
  float v = (x[i] - mu) * sc;
  x[i] = v > 0.0f ? v : 0.0f;
}

__global__ void pool_sum(const float* __restrict__ x, const int* __restrict__ batch,
                         float* __restrict__ pooled, int n_rows) {
  int f = threadIdx.x;                        // blockDim = 128
  for (int r = blockIdx.x; r < n_rows; r += gridDim.x) {
    int g = batch[r];
    atomicAdd(&pooled[g * DIM + f], x[(size_t)r * DIM + f]);
  }
}

__global__ void pool_count(const int* __restrict__ batch, float* counts, int n_rows) {
  int i = blockIdx.x * blockDim.x + threadIdx.x;
  if (i < n_rows) atomicAdd(&counts[batch[i]], 1.0f);
}

__global__ void fc_kernel(const float* __restrict__ pooled, const float* __restrict__ counts,
                          const float* __restrict__ fw, const float* __restrict__ fb,
                          float* __restrict__ out) {
  int i = blockIdx.x * blockDim.x + threadIdx.x;   // 64*10
  if (i >= NG * NC) return;
  int g = i / NC, c = i % NC;
  float cnt = counts[g]; cnt = cnt > 1.0f ? cnt : 1.0f;
  float inv = 1.0f / cnt;
  float acc = fb[c];
  for (int k = 0; k < DIM; ++k)
    acc += (pooled[g * DIM + k] * inv) * fw[k * NC + c];
  out[i] = acc;
}

extern "C" void kernel_launch(void* const* d_in, const int* in_sizes, int n_in,
                              void* d_out, int out_size, void* d_ws, size_t ws_size,
                              hipStream_t stream) {
  (void)in_sizes; (void)n_in; (void)out_size; (void)ws_size;

  const float* x      = (const float*)d_in[0];
  const int*   ei     = (const int*)d_in[1];
  const int*   batch  = (const int*)d_in[2];
  const float* lin_w  = (const float*)d_in[3];
  const float* lin_b  = (const float*)d_in[4];
  const float* out1_w = (const float*)d_in[5];
  const float* out1_b = (const float*)d_in[6];
  const float* out2_w = (const float*)d_in[7];
  const float* out2_b = (const float*)d_in[8];
  const float* fc_w   = (const float*)d_in[9];
  const float* fc_b   = (const float*)d_in[10];
  float* out = (float*)d_out;

  const size_t nd = (size_t)NN * DIM;         // 6.4M floats
  float* h      = (float*)d_ws;
  float* agg    = h + nd;
  float* xn     = agg + nd;
  float* sums   = xn + nd;                    // 128
  float* sumsq  = sums + 128;                 // 128
  float* pooled = sumsq + 128;                // 64*128
  float* counts = pooled + NG * DIM;          // 64

  const int gemm_blocks = (NN + 63) / 64;     // 782
  const int edge_blocks = (int)(((long long)NE * 32 + 255) / 256);
  const int elem_blocks = (int)((nd + 255) / 256);

  for (int i = 0; i < 5; ++i) {
    const float* Ain = (i == 0) ? x : xn;
    const size_t wofs = (size_t)i * DIM * DIM;
    const size_t bofs = (size_t)i * DIM;

    // h = Ain @ lin_w[i] + lin_b[i]
    gemm128<<<gemm_blocks, 256, 0, stream>>>(Ain, lin_w + wofs, lin_b + bofs, h, NN, 0);
    // agg = h (self loop), then scatter-add over edges
    hipMemcpyAsync(agg, h, nd * sizeof(float), hipMemcpyDeviceToDevice, stream);
    edge_scatter<<<edge_blocks, 256, 0, stream>>>(h, ei, agg, NE);
    // h = relu(agg) @ out1_w[i] + out1_b[i]
    gemm128<<<gemm_blocks, 256, 0, stream>>>(agg, out1_w + wofs, out1_b + bofs, h, NN, 1);
    // xn = relu(h) @ out2_w[i] + out2_b[i]
    gemm128<<<gemm_blocks, 256, 0, stream>>>(h, out2_w + wofs, out2_b + bofs, xn, NN, 1);

    if (i < 4) {
      zero_f32<<<1, 256, 0, stream>>>(sums, 256);           // sums + sumsq contiguous
      bn_stats<<<512, 128, 0, stream>>>(xn, sums, sumsq, NN);
      bn_apply_relu<<<elem_blocks, 256, 0, stream>>>(xn, sums, sumsq, NN);
    }
  }

  zero_f32<<<(NG * DIM + NG + 255) / 256, 256, 0, stream>>>(pooled, NG * DIM + NG);
  pool_sum<<<512, 128, 0, stream>>>(xn, batch, pooled, NN);
  pool_count<<<(NN + 255) / 256, 256, 0, stream>>>(batch, counts, NN);
  fc_kernel<<<(NG * NC + 255) / 256, 256, 0, stream>>>(pooled, counts, fc_w, fc_b, out);
}
